// CirLinear_33225867002303
// MI455X (gfx1250) — matrix operand order repaired and checked by reading, not compiled, not run on hardware
//
#include <hip/hip_runtime.h>
#include <hip/hip_bf16.h>

// ---------------------------------------------------------------------------
// CirLinear on MI455X (gfx1250): weight projection + bf16 WMMA GEMM
//   x:      [16,512,1024] f32   -> packed bf16 A tiles (WMMA lane layout)
//   weight: [4096,1024]   f32   -> circulant-mixed, packed bf16 B tiles
//   out:    [16,512,4096] f32 = x @ W_mix^T + bias
//
// GEMM: block tile 256(M) x 64(N); 8 waves, each wave 32(M) x 64(N).
// B tiles (4 KB / K-step) staged to LDS once per block via
// global_load_async_to_lds_b128 (double-buffered, drained with
// s_wait_asynccnt), shared by all 8 waves.
// ---------------------------------------------------------------------------

typedef __attribute__((ext_vector_type(16))) __bf16        v16bf;
typedef __attribute__((ext_vector_type(8)))  float         v8f;
typedef __attribute__((ext_vector_type(4)))  unsigned int  v4u;

// exact pointee type of the async-LDS builtin (from clang's diagnostic):
typedef int cir_v4i __attribute__((vector_size(4 * sizeof(int))));
typedef __attribute__((address_space(1))) cir_v4i* cir_gp1;   // global (AS1)
typedef __attribute__((address_space(3))) cir_v4i* cir_lp3;   // LDS    (AS3)

#define IN_F   1024
#define OUT_F  4096
#define M_TOT  8192            // 16*512
#define NKSTEP (IN_F / 32)     // 32 K-steps of K=32

#if defined(__has_builtin)
#if __has_builtin(__builtin_amdgcn_global_load_async_to_lds_b128)
#define CIR_ASYNC_LDS 1
#endif
#endif

__device__ __forceinline__ unsigned short f32_to_bf16_rne(float v) {
    unsigned int u = __float_as_uint(v);
    unsigned int r = (u + 0x7FFFu + ((u >> 16) & 1u)) >> 16;
    return (unsigned short)r;
}

// --------------------------------------------------------------------------
// Pack x (f32) into bf16 WMMA A-matrix tile layout.
// Tile = 16(M) x 32(K) = 512 halves = 1024 B, per-lane 32 B:
//   lane = mlocal + 16*((klocal/8)&1);  half = (klocal%8) + 8*(klocal/16)
// --------------------------------------------------------------------------
__global__ void cir_pack_x(const float* __restrict__ x,
                           unsigned short* __restrict__ xp) {
    int idx = blockIdx.x * 256 + threadIdx.x;     // 0 .. 8M-1
    int m = idx >> 10;                             // / IN_F
    int k = idx & (IN_F - 1);
    unsigned short b = f32_to_bf16_rne(x[idx]);
    int mt = m >> 4, ml = m & 15;
    int kt = k >> 5, kl = k & 31;
    int lane = ml + 16 * ((kl >> 3) & 1);
    int h    = (kl & 7) + 8 * (kl >> 4);
    size_t off = (((size_t)mt * NKSTEP + kt) << 9) + lane * 16 + h;
    xp[off] = b;
}

// --------------------------------------------------------------------------
// Build W_mix = a0*W + sum_p a_p * circ_avg(W, 2^p), one 16x16 block per CTA.
// Emit bf16 into WMMA B-matrix tile layout (K=32 x N=16 tiles):
//   lane = nlocal + 16*((klocal/8)&1);  half = (klocal%8) + 8*(klocal/16)
// --------------------------------------------------------------------------
__global__ void cir_build_w(const float* __restrict__ weight,
                            const float* __restrict__ alphas,
                            unsigned short* __restrict__ wp) {
    __shared__ float sw[16][17];                  // padded: no bank conflicts
    int t  = threadIdx.x;
    int ol = t >> 4, il = t & 15;
    int ot = blockIdx.x >> 6;                     // 0..255  (4096/16)
    int it = blockIdx.x & 63;                     // 0..63   (1024/16)
    int o0 = ot << 4, i0 = it << 4;

    sw[ol][il] = weight[(size_t)(o0 + ol) * IN_F + i0 + il];

    // softmax over 5 alphas (computed redundantly per thread; trivial cost)
    float a0 = alphas[0], a1 = alphas[1], a2 = alphas[2],
          a3 = alphas[3], a4 = alphas[4];
    float mx = fmaxf(fmaxf(fmaxf(a0, a1), fmaxf(a2, a3)), a4);
    float e0 = __expf(a0 - mx), e1 = __expf(a1 - mx), e2 = __expf(a2 - mx),
          e3 = __expf(a3 - mx), e4 = __expf(a4 - mx);
    float inv = 1.0f / (e0 + e1 + e2 + e3 + e4);
    float aw[5] = { e0 * inv, e1 * inv, e2 * inv, e3 * inv, e4 * inv };

    __syncthreads();

    float acc = aw[0] * sw[ol][il];
#pragma unroll
    for (int p = 1; p <= 4; ++p) {
        int bs  = 1 << p;
        int msk = bs - 1;
        int ob  = ol & ~msk, ib = il & ~msk;
        int d   = (ol - il) & msk;
        float s = 0.0f;
#pragma unroll
        for (int j = 0; j < 16; ++j) {            // max bs; guarded
            if (j < bs) s += sw[ob + ((j + d) & msk)][ib + j];
        }
        acc += aw[p] * s * (1.0f / (float)bs);
    }

    unsigned short b = f32_to_bf16_rne(acc);
    int n = o0 + ol;                              // output-feature index (N)
    int k = i0 + il;                              // input-feature index  (K)
    int kt = k >> 5, kl = k & 31;
    int nt = n >> 4, nl = n & 15;
    int lane = nl + 16 * ((kl >> 3) & 1);
    int h    = (kl & 7) + 8 * (kl >> 4);
    wp[(((size_t)kt * (OUT_F / 16) + nt) << 9) + lane * 16 + h] = b;
}

// --------------------------------------------------------------------------
// GEMM with LDS-shared, double-buffered B tiles (async LDS copy).
// --------------------------------------------------------------------------
__global__ void __launch_bounds__(256)
cir_gemm(const unsigned short* __restrict__ xp,
         const unsigned short* __restrict__ wp,
         const float* __restrict__ bias,
         float* __restrict__ out) {
    __shared__ __align__(128) unsigned short bsm[2][2048];   // 2 x 4 KB

    const int tid  = threadIdx.x;
    const int lane = tid & 31;
    const int wv   = tid >> 5;
    const int mblk = blockIdx.x >> 6;             // 0..31  (8192/256)
    const int ngrp = blockIdx.x & 63;             // 0..63  (4096/64)
    const int nt0  = ngrp << 2;                   // 4 shared n-tiles
    const int mt0  = (mblk << 4) + (wv << 1);     // wave's 2 m-tiles

    v8f c00 = {}, c01 = {}, c02 = {}, c03 = {};
    v8f c10 = {}, c11 = {}, c12 = {}, c13 = {};

    // --- stage one 4 KB B chunk (4 tiles) into LDS buffer `buf` ---
#ifdef CIR_ASYNC_LDS
#define CIR_STAGE_IMPL(SRC, DST)                                              \
    __builtin_amdgcn_global_load_async_to_lds_b128((cir_gp1)(SRC),            \
                                                   (cir_lp3)(DST), 0, 0);
#define CIR_WAIT_ASYNC() asm volatile("s_wait_asynccnt 0x0" ::: "memory")
#else
#define CIR_STAGE_IMPL(SRC, DST) *(v4u*)(DST) = *(const v4u*)(SRC);
#define CIR_WAIT_ASYNC()
#endif
#define CIR_STAGE(kt, buf)                                                    \
    {                                                                         \
        const unsigned short* _src =                                          \
            wp + (((size_t)(kt) * (OUT_F / 16) + nt0) << 9) + tid * 8;        \
        unsigned short* _dst = &bsm[(buf)][tid * 8];                          \
        CIR_STAGE_IMPL(_src, _dst)                                            \
    }

    CIR_STAGE(0, 0)

    for (int kt = 0; kt < NKSTEP; ++kt) {
        CIR_WAIT_ASYNC();          // B chunk for step kt has landed in LDS
        __syncthreads();           // ... and is visible to all waves
        if (kt + 1 < NKSTEP)       // overlap next copy with this step's math
            CIR_STAGE(kt + 1, (kt + 1) & 1)

        // A: 2 tiles for this wave (contiguous 32 B per lane)
        size_t abase = ((size_t)mt0 * NKSTEP + kt) << 9;
        v16bf a0 = ((const v16bf*)(xp + abase))[lane];
        v16bf a1 = ((const v16bf*)(xp + abase + ((size_t)NKSTEP << 9)))[lane];
        if (kt + 2 < NKSTEP)
            __builtin_prefetch(xp + abase + 2048, 0, 1);

        // B: 4 tiles from LDS (shared across the 8 waves)
        const v16bf* Bt = (const v16bf*)&bsm[kt & 1][0];
        v16bf b0 = Bt[lane];
        v16bf b1 = Bt[32 + lane];
        v16bf b2 = Bt[64 + lane];
        v16bf b3 = Bt[96 + lane];

        c00 = __builtin_amdgcn_wmma_f32_16x16x32_bf16(false, a0, false, b0,
                                                      (short)0, c00, false, false);
        c10 = __builtin_amdgcn_wmma_f32_16x16x32_bf16(false, a1, false, b0,
                                                      (short)0, c10, false, false);
        c01 = __builtin_amdgcn_wmma_f32_16x16x32_bf16(false, a0, false, b1,
                                                      (short)0, c01, false, false);
        c11 = __builtin_amdgcn_wmma_f32_16x16x32_bf16(false, a1, false, b1,
                                                      (short)0, c11, false, false);
        c02 = __builtin_amdgcn_wmma_f32_16x16x32_bf16(false, a0, false, b2,
                                                      (short)0, c02, false, false);
        c12 = __builtin_amdgcn_wmma_f32_16x16x32_bf16(false, a1, false, b2,
                                                      (short)0, c12, false, false);
        c03 = __builtin_amdgcn_wmma_f32_16x16x32_bf16(false, a0, false, b3,
                                                      (short)0, c03, false, false);
        c13 = __builtin_amdgcn_wmma_f32_16x16x32_bf16(false, a1, false, b3,
                                                      (short)0, c13, false, false);
    }

    // C/D layout: vgpr r -> M = r + 8*(lane/16), N = lane%16
    const int nl = lane & 15;
    const int mo = (lane >> 4) << 3;

#define CIR_STORE(ci, mtile, i)                                              \
    {                                                                        \
        int n  = (nt0 << 4) + ((i) << 4) + nl;                               \
        int m0 = ((mtile) << 4) + mo;                                        \
        float bv = bias[n];                                                  \
        _Pragma("unroll")                                                    \
        for (int r = 0; r < 8; ++r)                                          \
            out[(size_t)(m0 + r) * OUT_F + n] = (ci)[r] + bv;                \
    }
    CIR_STORE(c00, mt0,     0)
    CIR_STORE(c01, mt0,     1)
    CIR_STORE(c02, mt0,     2)
    CIR_STORE(c03, mt0,     3)
    CIR_STORE(c10, mt0 + 1, 0)
    CIR_STORE(c11, mt0 + 1, 1)
    CIR_STORE(c12, mt0 + 1, 2)
    CIR_STORE(c13, mt0 + 1, 3)
#undef CIR_STORE
#undef CIR_STAGE
#undef CIR_STAGE_IMPL
#undef CIR_WAIT_ASYNC
}

// --------------------------------------------------------------------------
extern "C" void kernel_launch(void* const* d_in, const int* in_sizes, int n_in,
                              void* d_out, int out_size, void* d_ws, size_t ws_size,
                              hipStream_t stream) {
    const float* x      = (const float*)d_in[0];
    const float* weight = (const float*)d_in[1];
    const float* alphas = (const float*)d_in[2];
    const float* bias   = (const float*)d_in[3];
    float*       out    = (float*)d_out;

    unsigned short* xp = (unsigned short*)d_ws;                 // 16 MB
    unsigned short* wp = xp + (size_t)M_TOT * IN_F;             // +8 MB

    // 1) pack x -> bf16 A tiles
    cir_pack_x<<<(M_TOT * IN_F) / 256, 256, 0, stream>>>(x, xp);

    // 2) circulant-mix weight -> bf16 B tiles
    cir_build_w<<<(OUT_F / 16) * (IN_F / 16), 256, 0, stream>>>(weight, alphas, wp);

    // 3) WMMA GEMM + bias (256x64 block tile, LDS-shared B)
    cir_gemm<<<(M_TOT / 256) * (OUT_F / 64), 256, 0, stream>>>(xp, wp, bias, out);
}
